// MambaTransformerBlock_53137335386572
// MI455X (gfx1250) — matrix operand order repaired and checked by reading, not compiled, not run on hardware
//
#include <hip/hip_runtime.h>
#include <math.h>

// ---------------------------------------------------------------------------
// MI455X / gfx1250 MambaTransformer block.
// All dense layers: v_wmma_f32_16x16x32_bf16, f32->bf16 packed as uint32 at
// staging, LDS strides chosen so every WMMA fragment is 2x ds_load_b128.
// GEMM: 256 thr / 8 waves, 128x64 tile, double-buffered LDS, float4 global
// staging. Attention: flash-style, K/V cooperatively staged per block
// (V transposed in LDS for vector B-fragments). Workspace: 9 x 9216*768 f32.
// ---------------------------------------------------------------------------

#define HIDDEN 768
#define HEADS  12
#define DH     64
#define MLPD   3072
#define DSTATE 16
#define DTRANK 48
#define BBATCH 16
#define HHH    24
#define WWW    24
#define LLEN   (HHH * WWW)           // 576
#define NTOK   (BBATCH * LLEN)       // 9216
#define XPN    (DTRANK + 2 * DSTATE) // 80
#define NSEQ   (BBATCH * HHH)        // 384

#define GBM 128
#define GBN 64
#define LDSR 40   // GEMM/P LDS row stride (ushorts): 16B-aligns b128 frags
#define KLDR 72   // flash K-tile row stride (64 dh + 8 pad)

typedef __attribute__((ext_vector_type(16))) __bf16   v16bf;
typedef __attribute__((ext_vector_type(8)))  float    v8f;
typedef __attribute__((ext_vector_type(8)))  unsigned v8u;
typedef __attribute__((ext_vector_type(4)))  unsigned v4u;

// float -> bf16 bits, round-to-nearest-even
static __device__ __forceinline__ unsigned short f2bf(float f) {
  unsigned u = __builtin_bit_cast(unsigned, f);
  u += 0x7FFFu + ((u >> 16) & 1u);
  return (unsigned short)(u >> 16);
}
// pack two floats into one dword of two bf16
static __device__ __forceinline__ unsigned pk2(float a, float b) {
  return (unsigned)f2bf(a) | ((unsigned)f2bf(b) << 16);
}
static __device__ __forceinline__ v16bf frag_cat(v4u lo, v4u hi) {
  v8u u;
  u[0] = lo[0]; u[1] = lo[1]; u[2] = lo[2]; u[3] = lo[3];
  u[4] = hi[0]; u[5] = hi[1]; u[6] = hi[2]; u[7] = hi[3];
  return __builtin_bit_cast(v16bf, u);
}
static __device__ __forceinline__ float gelu_f(float x) {
  return 0.5f * x * (1.0f + erff(x * 0.70710678118654752f));
}
static __device__ __forceinline__ float softplus_f(float x) {
  return (x > 30.f) ? x : log1pf(__expf(x));
}
static __device__ __forceinline__ float silu_f(float x) {
  return x / (1.f + __expf(-x));
}
static __device__ __forceinline__ float wred_max16(float v) {
  v = fmaxf(v, __shfl_xor(v, 1, 32));
  v = fmaxf(v, __shfl_xor(v, 2, 32));
  v = fmaxf(v, __shfl_xor(v, 4, 32));
  v = fmaxf(v, __shfl_xor(v, 8, 32));
  return v;
}
static __device__ __forceinline__ float wred_sum16(float v) {
  v += __shfl_xor(v, 1, 32);
  v += __shfl_xor(v, 2, 32);
  v += __shfl_xor(v, 4, 32);
  v += __shfl_xor(v, 8, 32);
  return v;
}

// ---------------------------------------------------------------------------
// LayerNorm: one block per token, C=768
// ---------------------------------------------------------------------------
__global__ __launch_bounds__(256)
void layernorm_k(const float* __restrict__ X, const float* __restrict__ w,
                 const float* __restrict__ b, float* __restrict__ Y) {
  __shared__ float red[256];
  const int row = blockIdx.x;
  const float* x = X + (long)row * HIDDEN;
  float s = 0.f;
  for (int c = threadIdx.x; c < HIDDEN; c += 256) s += x[c];
  red[threadIdx.x] = s;
  __syncthreads();
  for (int off = 128; off > 0; off >>= 1) {
    if (threadIdx.x < off) red[threadIdx.x] += red[threadIdx.x + off];
    __syncthreads();
  }
  const float mean = red[0] * (1.f / HIDDEN);
  __syncthreads();
  float s2 = 0.f;
  for (int c = threadIdx.x; c < HIDDEN; c += 256) {
    float d = x[c] - mean;
    s2 += d * d;
  }
  red[threadIdx.x] = s2;
  __syncthreads();
  for (int off = 128; off > 0; off >>= 1) {
    if (threadIdx.x < off) red[threadIdx.x] += red[threadIdx.x + off];
    __syncthreads();
  }
  const float inv = rsqrtf(red[0] * (1.f / HIDDEN) + 1e-6f);
  for (int c = threadIdx.x; c < HIDDEN; c += 256)
    Y[(long)row * HIDDEN + c] = (x[c] - mean) * inv * w[c] + b[c];
}

// ---------------------------------------------------------------------------
// WMMA GEMM: D = act(A[M,K] @ W[N,K]^T + bias) (+resid)
// 256 threads / 8 waves, 128x64 tile, K step 32, double-buffered LDS.
// ACT: 0=none 1=gelu 2=softplus
// ---------------------------------------------------------------------------
template <int ACT, bool RESID>
__global__ __launch_bounds__(256)
void gemm_wmma_k(const float* __restrict__ A, int lda,
                 const float* __restrict__ W, int ldw,
                 const float* __restrict__ bias,
                 const float* __restrict__ resid,
                 float* __restrict__ D, int ldd, int M, int N, int K) {
  __shared__ alignas(16) unsigned short As[2][GBM * LDSR];
  __shared__ alignas(16) unsigned short Ws[2][GBN * LDSR];
  const int tid  = threadIdx.x;
  const int wave = tid >> 5;
  const int lane = tid & 31;
  const int half = lane >> 4;
  const int l16  = lane & 15;
  const int m0 = blockIdx.y * GBM;
  const int n0 = blockIdx.x * GBN;

  float4 ra[4], rw[2];

  auto gload = [&](int k0) {
#pragma unroll
    for (int j = 0; j < 4; ++j) {           // A tile: 128x32 = 1024 float4
      const int idx = tid + j * 256;
      const int r = idx >> 3, c = (idx & 7) << 2;
      const int m = m0 + r, k = k0 + c;
      if (m < M && k + 3 < K) {
        ra[j] = *(const float4*)(A + (long)m * lda + k);
      } else {
        float v0 = (m < M && k + 0 < K) ? A[(long)m * lda + k + 0] : 0.f;
        float v1 = (m < M && k + 1 < K) ? A[(long)m * lda + k + 1] : 0.f;
        float v2 = (m < M && k + 2 < K) ? A[(long)m * lda + k + 2] : 0.f;
        float v3 = (m < M && k + 3 < K) ? A[(long)m * lda + k + 3] : 0.f;
        ra[j] = make_float4(v0, v1, v2, v3);
      }
    }
#pragma unroll
    for (int j = 0; j < 2; ++j) {           // W tile: 64x32 = 512 float4
      const int idx = tid + j * 256;
      const int r = idx >> 3, c = (idx & 7) << 2;
      const int n = n0 + r, k = k0 + c;
      if (n < N && k + 3 < K) {
        rw[j] = *(const float4*)(W + (long)n * ldw + k);
      } else {
        float v0 = (n < N && k + 0 < K) ? W[(long)n * ldw + k + 0] : 0.f;
        float v1 = (n < N && k + 1 < K) ? W[(long)n * ldw + k + 1] : 0.f;
        float v2 = (n < N && k + 2 < K) ? W[(long)n * ldw + k + 2] : 0.f;
        float v3 = (n < N && k + 3 < K) ? W[(long)n * ldw + k + 3] : 0.f;
        rw[j] = make_float4(v0, v1, v2, v3);
      }
    }
  };
  auto sstore = [&](int buf) {
#pragma unroll
    for (int j = 0; j < 4; ++j) {
      const int idx = tid + j * 256;
      const int r = idx >> 3, c = (idx & 7) << 2;
      uint2 p;
      p.x = pk2(ra[j].x, ra[j].y);
      p.y = pk2(ra[j].z, ra[j].w);
      *(uint2*)&As[buf][r * LDSR + c] = p;
    }
#pragma unroll
    for (int j = 0; j < 2; ++j) {
      const int idx = tid + j * 256;
      const int r = idx >> 3, c = (idx & 7) << 2;
      uint2 p;
      p.x = pk2(rw[j].x, rw[j].y);
      p.y = pk2(rw[j].z, rw[j].w);
      *(uint2*)&Ws[buf][r * LDSR + c] = p;
    }
  };

  v8f acc0 = {0.f,0.f,0.f,0.f,0.f,0.f,0.f,0.f};
  v8f acc1 = acc0, acc2 = acc0, acc3 = acc0;

  gload(0);
  sstore(0);
  __syncthreads();
  int buf = 0;
  for (int k0 = 0; k0 < K; k0 += 32) {
    const bool last = (k0 + 32 >= K);
    if (!last) gload(k0 + 32);
    // A fragment: row = wave*16+l16; k = e + 8*half (e<8), 16+e-8+8*half (e>=8)
    const unsigned short* arow = &As[buf][(wave * 16 + l16) * LDSR + (half << 3)];
    const v16bf a = frag_cat(*(const v4u*)arow, *(const v4u*)(arow + 16));
    // B fragments: col = l16; k = e + 8*half (e<8), 16+e-8+8*half (e>=8)
#pragma unroll
    for (int nt = 0; nt < 4; ++nt) {
      const unsigned short* brow = &Ws[buf][(nt * 16 + l16) * LDSR + (half << 3)];
      const v16bf bfr = frag_cat(*(const v4u*)brow, *(const v4u*)(brow + 16));
      v8f* accp = (nt == 0) ? &acc0 : (nt == 1) ? &acc1 : (nt == 2) ? &acc2 : &acc3;
      *accp = __builtin_amdgcn_wmma_f32_16x16x32_bf16(false, a, false, bfr,
                                                      (short)0, *accp, false, false);
    }
    if (last) break;
    sstore(buf ^ 1);
    __syncthreads();
    buf ^= 1;
  }

  v8f accs[4] = {acc0, acc1, acc2, acc3};
#pragma unroll
  for (int nt = 0; nt < 4; ++nt) {
    const int n = n0 + nt * 16 + l16;
    if (n >= N) continue;
    const float bv = bias ? bias[n] : 0.f;
#pragma unroll
    for (int r = 0; r < 8; ++r) {
      const int m = m0 + wave * 16 + (half << 3) + r;
      if (m >= M) continue;
      float v = accs[nt][r] + bv;
      if (ACT == 1) v = gelu_f(v);
      else if (ACT == 2) v = softplus_f(v);
      if (RESID) v += resid[(long)m * ldd + n];
      D[(long)m * ldd + n] = v;
    }
  }
}

// ---------------------------------------------------------------------------
// Depthwise causal conv (4 taps) + bias + SiLU along the 24-step axis.
// ---------------------------------------------------------------------------
__global__ __launch_bounds__(256)
void conv_silu_k(const float* __restrict__ xz, const float* __restrict__ cw,
                 const float* __restrict__ cb, float* __restrict__ out) {
  const int idx = blockIdx.x * 256 + threadIdx.x;
  if (idx >= NSEQ * HIDDEN) return;
  const int d = idx % HIDDEN, s = idx / HIDDEN;
  const float w0 = cw[d * 4 + 0], w1 = cw[d * 4 + 1];
  const float w2 = cw[d * 4 + 2], w3 = cw[d * 4 + 3];
  const float bv = cb[d];
  const float* xr = xz + (long)s * WWW * 2 * HIDDEN + d;
  float h0 = 0.f, h1 = 0.f, h2 = 0.f;
  for (int t = 0; t < WWW; ++t) {
    const float xt = xr[(long)t * 2 * HIDDEN];
    const float a = bv + w3 * xt + w2 * h0 + w1 * h1 + w0 * h2;
    out[((long)s * WWW + t) * HIDDEN + d] = silu_f(a);
    h2 = h1; h1 = h0; h0 = xt;
  }
}

// ---------------------------------------------------------------------------
// Selective scan + skip + SiLU gate.
// ---------------------------------------------------------------------------
__global__ __launch_bounds__(256)
void scan_k(const float* __restrict__ dt, const float* __restrict__ xc,
            const float* __restrict__ xdbl, const float* __restrict__ xz,
            const float* __restrict__ A_log, const float* __restrict__ Dp,
            float* __restrict__ y) {
  const int idx = blockIdx.x * 256 + threadIdx.x;
  if (idx >= NSEQ * HIDDEN) return;
  const int d = idx % HIDDEN, s = idx / HIDDEN;
  float Av[DSTATE], h[DSTATE];
#pragma unroll
  for (int n = 0; n < DSTATE; ++n) {
    Av[n] = -__expf(A_log[d * DSTATE + n]);
    h[n] = 0.f;
  }
  const float Dv = Dp[d];
  for (int t = 0; t < WWW; ++t) {
    const long row = (long)s * WWW + t;
    const float dtv = dt[row * HIDDEN + d];
    const float xv = xc[row * HIDDEN + d];
    const float* bd = xdbl + row * XPN;
    float yv = 0.f;
#pragma unroll
    for (int n = 0; n < DSTATE; ++n) {
      const float Bn = bd[DTRANK + n];
      const float Cn = bd[DTRANK + DSTATE + n];
      h[n] = __expf(dtv * Av[n]) * h[n] + dtv * xv * Bn;
      yv += h[n] * Cn;
    }
    yv += xv * Dv;
    const float z = xz[row * 2 * HIDDEN + HIDDEN + d];
    y[row * HIDDEN + d] = yv * silu_f(z);
  }
}

// ---------------------------------------------------------------------------
// Permutations + fused gather-multiply
// ---------------------------------------------------------------------------
__global__ __launch_bounds__(256)
void perm_h_k(const float* __restrict__ src, float* __restrict__ dst) {
  const long id = (long)blockIdx.x * 256 + threadIdx.x;
  if (id >= (long)NTOK * HIDDEN) return;
  const int c = (int)(id % HIDDEN);
  const int t = (int)((id / HIDDEN) % WWW);
  const int s = (int)(id / ((long)HIDDEN * WWW));
  const int w = s % WWW, b = s / WWW;
  dst[id] = src[(((long)b * HHH + t) * WWW + w) * HIDDEN + c];
}
__global__ __launch_bounds__(256)
void perm_v_k(const float* __restrict__ src, float* __restrict__ dst) {
  // dst is flat (B, H, C, W) order (transpose (0,1,3,2) of (B,H,W,C))
  const long id = (long)blockIdx.x * 256 + threadIdx.x;
  if (id >= (long)NTOK * HIDDEN) return;
  const int w  = (int)(id % WWW);
  const int c  = (int)((id / WWW) % HIDDEN);
  const int hh = (int)((id / ((long)WWW * HIDDEN)) % HHH);
  const int b  = (int)(id / ((long)WWW * HIDDEN * HHH));
  dst[id] = src[(((long)b * HHH + hh) * WWW + w) * HIDDEN + c];
}
__global__ __launch_bounds__(256)
void fmul_k(const float* __restrict__ mh, const float* __restrict__ mv,
            float* __restrict__ dst) {
  const long i = (long)blockIdx.x * 256 + threadIdx.x;
  if (i >= (long)NTOK * HIDDEN) return;
  const int c  = (int)(i % HIDDEN);
  const long tok = i / HIDDEN;
  const int w  = (int)(tok % WWW);
  const int hh = (int)((tok / WWW) % HHH);
  const int b  = (int)(tok / LLEN);
  const float xh = mh[(((long)b * WWW + w) * HHH + hh) * HIDDEN + c];
  dst[i] = xh * mv[i];
}

// ---------------------------------------------------------------------------
// Flash attention: block = 128 thr / 4 waves, 64 queries per block,
// K/V (32 keys x 64 dh) cooperatively staged per chunk; V transposed in LDS.
// ---------------------------------------------------------------------------
__global__ __launch_bounds__(128)
void flash_attn_k(const float* __restrict__ Q, const float* __restrict__ K,
                  const float* __restrict__ V, float* __restrict__ O) {
  __shared__ alignas(16) unsigned short Kl[32 * KLDR];      // [key][dh]
  __shared__ alignas(16) unsigned short Vt[64 * LDSR];      // [dh][key]
  __shared__ alignas(16) unsigned short Pl[4][16 * LDSR];   // per wave [q][key]
  const int tid  = threadIdx.x;
  const int wave = tid >> 5, lane = tid & 31;
  const int half = lane >> 4, l16 = lane & 15;
  const int bh = blockIdx.x;
  const int b = bh / HEADS, h = bh % HEADS;
  const int q0 = blockIdx.y * 64 + wave * 16;
  const long base = (long)b * LLEN * HIDDEN + h * DH;

  // Q fragments (vectorized): e<8 -> d = e + 8*half + 32*st; e>=8 -> +16
  v16bf aq[2];
  const float* qrow = Q + base + (long)(q0 + l16) * HIDDEN;
#pragma unroll
  for (int st = 0; st < 2; ++st) {
    const int d0 = (half << 3) + 32 * st;
    const float4 f0 = *(const float4*)(qrow + d0);
    const float4 f1 = *(const float4*)(qrow + d0 + 4);
    const float4 f2 = *(const float4*)(qrow + d0 + 16);
    const float4 f3 = *(const float4*)(qrow + d0 + 20);
    v8u u;
    u[0] = pk2(f0.x, f0.y); u[1] = pk2(f0.z, f0.w);
    u[2] = pk2(f1.x, f1.y); u[3] = pk2(f1.z, f1.w);
    u[4] = pk2(f2.x, f2.y); u[5] = pk2(f2.z, f2.w);
    u[6] = pk2(f3.x, f3.y); u[7] = pk2(f3.z, f3.w);
    aq[st] = __builtin_bit_cast(v16bf, u);
  }

  float mi[8], li[8];
#pragma unroll
  for (int r = 0; r < 8; ++r) { mi[r] = -1e30f; li[r] = 0.f; }
  v8f acc0 = {0.f,0.f,0.f,0.f,0.f,0.f,0.f,0.f};
  v8f acc1 = acc0, acc2 = acc0, acc3 = acc0;

  for (int kc = 0; kc < LLEN; kc += 32) {
    // cooperative staging: 32 keys x 64 dh = 512 float4, 128 threads -> 4 each
#pragma unroll
    for (int j = 0; j < 4; ++j) {
      const int idx = tid + j * 128;
      const int key = idx >> 4, c = (idx & 15) << 2;
      const float4 fk = *(const float4*)(K + base + (long)(kc + key) * HIDDEN + c);
      uint2 p;
      p.x = pk2(fk.x, fk.y);
      p.y = pk2(fk.z, fk.w);
      *(uint2*)&Kl[key * KLDR + c] = p;
      const float4 fv = *(const float4*)(V + base + (long)(kc + key) * HIDDEN + c);
      Vt[(c + 0) * LDSR + key] = f2bf(fv.x);
      Vt[(c + 1) * LDSR + key] = f2bf(fv.y);
      Vt[(c + 2) * LDSR + key] = f2bf(fv.z);
      Vt[(c + 3) * LDSR + key] = f2bf(fv.w);
    }
    __syncthreads();

    // S = Q K^T for two 16-key tiles
    v8f s0 = {0.f,0.f,0.f,0.f,0.f,0.f,0.f,0.f};
    v8f s1 = s0;
#pragma unroll
    for (int st = 0; st < 2; ++st) {
      const unsigned short* k0p = &Kl[(0 * 16 + l16) * KLDR + (half << 3) + 32 * st];
      const unsigned short* k1p = &Kl[(1 * 16 + l16) * KLDR + (half << 3) + 32 * st];
      const v16bf bk0 = frag_cat(*(const v4u*)k0p, *(const v4u*)(k0p + 16));
      const v16bf bk1 = frag_cat(*(const v4u*)k1p, *(const v4u*)(k1p + 16));
      s0 = __builtin_amdgcn_wmma_f32_16x16x32_bf16(false, aq[st], false, bk0, (short)0, s0, false, false);
      s1 = __builtin_amdgcn_wmma_f32_16x16x32_bf16(false, aq[st], false, bk1, (short)0, s1, false, false);
    }
    // online softmax (rows split across 16-lane halves)
#pragma unroll
    for (int r = 0; r < 8; ++r) {
      const float x0 = s0[r] * 0.125f, x1 = s1[r] * 0.125f;
      const float mx = wred_max16(fmaxf(x0, x1));
      const float mnew = fmaxf(mi[r], mx);
      const float corr = __expf(mi[r] - mnew);
      const float p0 = __expf(x0 - mnew), p1 = __expf(x1 - mnew);
      li[r] = li[r] * corr + wred_sum16(p0 + p1);
      mi[r] = mnew;
      acc0[r] *= corr; acc1[r] *= corr; acc2[r] *= corr; acc3[r] *= corr;
      const int row = r + (half << 3);
      Pl[wave][row * LDSR + l16]      = f2bf(p0);
      Pl[wave][row * LDSR + 16 + l16] = f2bf(p1);
    }
    asm volatile("s_wait_dscnt 0" ::: "memory");
    // P as A-fragment (vector reads)
    const unsigned short* prow = &Pl[wave][l16 * LDSR + (half << 3)];
    const v16bf pa = frag_cat(*(const v4u*)prow, *(const v4u*)(prow + 16));
    // P @ V from transposed V tile: col dh = nt*16+l16, k = e + 8*half (e<8), 16+e-8+8*half (e>=8)
#pragma unroll
    for (int nt = 0; nt < 4; ++nt) {
      const unsigned short* vrow = &Vt[(nt * 16 + l16) * LDSR + (half << 3)];
      const v16bf bv = frag_cat(*(const v4u*)vrow, *(const v4u*)(vrow + 16));
      v8f* accp = (nt == 0) ? &acc0 : (nt == 1) ? &acc1 : (nt == 2) ? &acc2 : &acc3;
      *accp = __builtin_amdgcn_wmma_f32_16x16x32_bf16(false, pa, false, bv, (short)0, *accp, false, false);
    }
    __syncthreads();  // before next chunk overwrites Kl/Vt
  }
#pragma unroll
  for (int r = 0; r < 8; ++r) {
    const float inv = 1.f / li[r];
    const int m = q0 + (half << 3) + r;
    float* op = O + ((long)b * LLEN + m) * HIDDEN + h * DH + l16;
    op[0]  = acc0[r] * inv;
    op[16] = acc1[r] * inv;
    op[32] = acc2[r] * inv;
    op[48] = acc3[r] * inv;
  }
}

// ---------------------------------------------------------------------------
// Host-side orchestration
// ---------------------------------------------------------------------------
template <int ACT, bool RESID>
static void gemm(const float* A, int lda, const float* W, int ldw,
                 const float* bias, const float* resid, float* D, int ldd,
                 int M, int N, int K, hipStream_t st) {
  dim3 g((N + GBN - 1) / GBN, (M + GBM - 1) / GBM), b(256);
  gemm_wmma_k<ACT, RESID><<<g, b, 0, st>>>(A, lda, W, ldw, bias, resid, D, ldd, M, N, K);
}

static void run_mamba(const float* xn, void* const* d_in, int pb, bool horiz,
                      float* r_perm, float* r_xz, float* r_conv, float* r_xdbl,
                      float* r_dt, float* r_y, float* out, hipStream_t st) {
  const long total = (long)NTOK * HIDDEN;
  dim3 eb(256), eg((unsigned)((total + 255) / 256));
  if (horiz) perm_h_k<<<eg, eb, 0, st>>>(xn, r_perm);
  else       perm_v_k<<<eg, eb, 0, st>>>(xn, r_perm);
  const float* in_w   = (const float*)d_in[pb + 0];
  const float* conv_w = (const float*)d_in[pb + 1];
  const float* conv_b = (const float*)d_in[pb + 2];
  const float* xp_w   = (const float*)d_in[pb + 3];
  const float* dt_w   = (const float*)d_in[pb + 4];
  const float* dt_b   = (const float*)d_in[pb + 5];
  const float* A_log  = (const float*)d_in[pb + 6];
  const float* Dp     = (const float*)d_in[pb + 7];
  const float* out_w  = (const float*)d_in[pb + 8];

  gemm<0, false>(r_perm, HIDDEN, in_w, HIDDEN, nullptr, nullptr,
                 r_xz, 2 * HIDDEN, NTOK, 2 * HIDDEN, HIDDEN, st);
  dim3 cg((NSEQ * HIDDEN + 255) / 256);
  conv_silu_k<<<cg, eb, 0, st>>>(r_xz, conv_w, conv_b, r_conv);
  gemm<0, false>(r_conv, HIDDEN, xp_w, HIDDEN, nullptr, nullptr,
                 r_xdbl, XPN, NTOK, XPN, HIDDEN, st);
  gemm<2, false>(r_xdbl, XPN, dt_w, DTRANK, dt_b, nullptr,
                 r_dt, HIDDEN, NTOK, HIDDEN, DTRANK, st);
  scan_k<<<cg, eb, 0, st>>>(r_dt, r_conv, r_xdbl, r_xz, A_log, Dp, r_y);
  gemm<0, false>(r_y, HIDDEN, out_w, HIDDEN, nullptr, nullptr,
                 out, HIDDEN, NTOK, HIDDEN, HIDDEN, st);
}

extern "C" void kernel_launch(void* const* d_in, const int* in_sizes, int n_in,
                              void* d_out, int out_size, void* d_ws, size_t ws_size,
                              hipStream_t stream) {
  (void)in_sizes; (void)n_in; (void)out_size; (void)ws_size;
  const float* x = (const float*)d_in[0];
  auto P = [&](int i) { return (const float*)d_in[i]; };
  float* R = (float*)d_ws;
  const long U = (long)NTOK * HIDDEN;
  auto r = [&](int i) { return R + (long)i * U; };

  dim3 eb(256), eg((unsigned)((U + 255) / 256));

  // xn = LN(x; mb_norm) -> r0
  layernorm_k<<<NTOK, 256, 0, stream>>>(x, P(1), P(2), r(0));

  // mamba horizontal -> r7, vertical -> r8
  run_mamba(r(0), d_in, 3, true,  r(1), r(2), r(4), r(1), r(5), r(6), r(7), stream);
  run_mamba(r(0), d_in, 12, false, r(1), r(2), r(4), r(1), r(5), r(6), r(8), stream);

  // fused = xh * xv -> r1 ; fus1 (gelu) -> r2 ; fus2 + x -> r3 (residual h)
  fmul_k<<<eg, eb, 0, stream>>>(r(7), r(8), r(1));
  gemm<1, false>(r(1), HIDDEN, P(21), HIDDEN, P(22), nullptr, r(2), HIDDEN,
                 NTOK, HIDDEN, HIDDEN, stream);
  gemm<0, true>(r(2), HIDDEN, P(23), HIDDEN, P(24), x, r(3), HIDDEN,
                NTOK, HIDDEN, HIDDEN, stream);

  // attention
  layernorm_k<<<NTOK, 256, 0, stream>>>(r(3), P(25), P(26), r(0));
  gemm<0, false>(r(0), HIDDEN, P(27), HIDDEN, P(28), nullptr, r(4), HIDDEN,
                 NTOK, HIDDEN, HIDDEN, stream);  // Q
  gemm<0, false>(r(0), HIDDEN, P(29), HIDDEN, P(30), nullptr, r(5), HIDDEN,
                 NTOK, HIDDEN, HIDDEN, stream);  // K
  gemm<0, false>(r(0), HIDDEN, P(31), HIDDEN, P(32), nullptr, r(6), HIDDEN,
                 NTOK, HIDDEN, HIDDEN, stream);  // V
  dim3 fg(BBATCH * HEADS, LLEN / 64), fb(128);
  flash_attn_k<<<fg, fb, 0, stream>>>(r(4), r(5), r(6), r(1));  // ctx -> r1
  gemm<0, true>(r(1), HIDDEN, P(33), HIDDEN, P(34), r(3), r(2), HIDDEN,
                NTOK, HIDDEN, HIDDEN, stream);   // o-proj + h

  // MLP
  layernorm_k<<<NTOK, 256, 0, stream>>>(r(2), P(35), P(36), r(0));
  gemm<1, false>(r(0), HIDDEN, P(37), HIDDEN, P(38), nullptr, r(4), MLPD,
                 NTOK, MLPD, HIDDEN, stream);            // fc1 gelu -> r4..r7
  gemm<0, true>(r(4), MLPD, P(39), MLPD, P(40), r(2), (float*)d_out, HIDDEN,
                NTOK, HIDDEN, MLPD, stream);             // fc2 + resid -> out
}
